// VectorQuantizerWithDiversity_8804682957159
// MI455X (gfx1250) — matrix operand order repaired and checked by reading, not compiled, not run on hardware
//
#include <hip/hip_runtime.h>
#include <hip/hip_bf16.h>

typedef __attribute__((ext_vector_type(16))) __bf16 v16bf;
typedef __attribute__((ext_vector_type(8)))  float  v8f;

#define K_CODES 1024
#define DIMS    256
#define ROWS_PB 16   // rows per block

// dynamic shared layout (bytes)
#define OFF_ZBF   0                         // 16*256*2  = 8192
#define OFF_ZF    (8192)                    // 16*256*4  = 16384
#define OFF_DIST  (8192 + 16384)            // 16*1024*4 = 65536
#define OFF_ZN    (8192 + 16384 + 65536)    // 16*4      = 64
#define SMEM_BYTES (8192 + 16384 + 65536 + 64)

// -------------------------------------------------------------------------
// Kernel 0: embed f32 -> bf16 + ||e||^2 per code. One wave per code.
// -------------------------------------------------------------------------
__global__ void __launch_bounds__(256)
vq_prep(const float* __restrict__ embed, __bf16* __restrict__ ebf,
        float* __restrict__ enorm)
{
    const int wv   = threadIdx.x >> 5;
    const int lane = threadIdx.x & 31;
    const int code = blockIdx.x * 8 + wv;
    const float* src = embed + (size_t)code * DIMS;
    float sq = 0.0f;
    for (int d = lane; d < DIMS; d += 32) {
        float v = src[d];
        ebf[(size_t)code * DIMS + d] = (__bf16)v;
        sq += v * v;
    }
    #pragma unroll
    for (int off = 16; off >= 1; off >>= 1) sq += __shfl_xor(sq, off, 32);
    if (lane == 0) enorm[code] = sq;
}

// -------------------------------------------------------------------------
// Kernel 1: main. 1 workgroup (8 waves) per 16-row tile of z.
//   GEMM: dist = ||z||^2 + ||e||^2 - 2 z.e^T via v_wmma_f32_16x16x32_bf16
//   then per-row argmin + softmax + gather + loss/usage accumulation.
// -------------------------------------------------------------------------
__global__ void __launch_bounds__(256)
vq_main(const float* __restrict__ z, const float* __restrict__ embed,
        const __bf16* __restrict__ ebf, const float* __restrict__ enorm,
        float* __restrict__ avgAcc, float* __restrict__ counts,
        float* __restrict__ lossAcc, float* __restrict__ out_zq,
        float* __restrict__ out_codes)
{
    extern __shared__ char smem[];
    __bf16* zbf   = (__bf16*)(smem + OFF_ZBF);
    float*  zf    = (float*)(smem + OFF_ZF);
    float*  dist  = (float*)(smem + OFF_DIST);
    float*  znorm = (float*)(smem + OFF_ZN);

    const int tid  = threadIdx.x;
    const int wv   = tid >> 5;       // wave 0..7
    const int lane = tid & 31;
    const int half = lane >> 4;      // 0 or 1
    const size_t rowBase = (size_t)blockIdx.x * ROWS_PB;

    if (tid < ROWS_PB) znorm[tid] = 0.0f;
    __syncthreads();

    // ---- Phase A: stage z tile (f32 + bf16) and row norms ----
    {
        const int r  = tid >> 4;             // 0..15
        const int c0 = (tid & 15) * 16;      // 16 floats per thread
        const float* src = z + (rowBase + r) * DIMS + c0;
        float sq = 0.0f;
        #pragma unroll
        for (int i = 0; i < 16; ++i) {
            float v = src[i];
            zf [r * DIMS + c0 + i] = v;
            zbf[r * DIMS + c0 + i] = (__bf16)v;
            sq += v * v;
        }
        atomicAdd(&znorm[r], sq);
    }
    __syncthreads();

    // ---- Phase B: WMMA GEMM. wave wv owns 8 N-tiles (128 codes). ----
    v8f acc[8];
    #pragma unroll
    for (int t = 0; t < 8; ++t)
        #pragma unroll
        for (int i = 0; i < 8; ++i) acc[t][i] = 0.0f;

    const int rowA = lane & 15;
    const __bf16* zrow = zbf + rowA * DIMS;

    #pragma unroll
    for (int k0 = 0; k0 < DIMS; k0 += 32) {
        // A fragment (16x32 bf16): contiguous k = 16*half + i, same order as B
        v16bf a;
        #pragma unroll
        for (int i = 0; i < 16; ++i) a[i]    = zrow[k0 + half * 16 + i];

        #pragma unroll
        for (int t = 0; t < 8; ++t) {
            const int code = (wv * 8 + t) * 16 + (lane & 15);
            // B fragment (32x16 bf16): 16 contiguous along D, 32B aligned
            const v16bf b = *(const v16bf*)(ebf + (size_t)code * DIMS + k0 + half * 16);
            acc[t] = __builtin_amdgcn_wmma_f32_16x16x32_bf16(
                false, a, false, b, (short)0, acc[t], false, false);
        }
    }

    // distances into LDS: dist[row][code] = ||z||^2 + ||e||^2 - 2*dot
    #pragma unroll
    for (int t = 0; t < 8; ++t) {
        const int col = (wv * 8 + t) * 16 + (lane & 15);
        const float en = enorm[col];
        #pragma unroll
        for (int i = 0; i < 8; ++i) {
            const int row = i + half * 8;   // C/D layout: VGPR i -> M=i / M=i+8
            dist[row * K_CODES + col] = znorm[row] + en - 2.0f * acc[t][i];
        }
    }
    __syncthreads();

    // ---- Phase C: per-row argmin + softmax (2 rows per wave) ----
    #pragma unroll
    for (int rr = 0; rr < 2; ++rr) {
        const int r = wv * 2 + rr;
        float* drow = dist + r * K_CODES;

        float minv = 3.402823e38f; int mini = 0;
        for (int k = lane; k < K_CODES; k += 32) {
            float dv = drow[k];
            if (dv < minv) { minv = dv; mini = k; }
        }
        #pragma unroll
        for (int off = 16; off >= 1; off >>= 1) {
            float om = __shfl_xor(minv, off, 32);
            int   oi = __shfl_xor(mini, off, 32);
            if (om < minv || (om == minv && oi < mini)) { minv = om; mini = oi; }
        }

        float s = 0.0f;
        for (int k = lane; k < K_CODES; k += 32) {
            float e = __expf(minv - drow[k]);   // T = 1.0
            drow[k] = e;
            s += e;
        }
        #pragma unroll
        for (int off = 16; off >= 1; off >>= 1) s += __shfl_xor(s, off, 32);
        const float invs = 1.0f / s;
        for (int k = lane; k < K_CODES; k += 32) drow[k] *= invs;

        // gather z_q, straight-through output, squared-error accumulation
        const int code = mini;
        const float* erow = embed + (size_t)code * DIMS;
        const float* zrowf = zf + r * DIMS;
        float* qdst = out_zq + (rowBase + r) * DIMS;
        float sq = 0.0f;
        for (int d = lane; d < DIMS; d += 32) {
            float q  = erow[d];
            float df = zrowf[d] - q;
            sq += df * df;
            qdst[d] = q;        // z + sg(z_q - z) == z_q numerically
        }
        #pragma unroll
        for (int off = 16; off >= 1; off >>= 1) sq += __shfl_xor(sq, off, 32);
        if (lane == 0) {
            out_codes[rowBase + r] = (float)code;
            atomicAdd(counts + code, 1.0f);
            atomicAdd(lossAcc, sq);
        }
    }
    __syncthreads();

    // ---- Phase D: column sums of probs -> global soft-usage accumulator ----
    for (int c = tid; c < K_CODES; c += 256) {
        float a = 0.0f;
        #pragma unroll
        for (int r = 0; r < ROWS_PB; ++r) a += dist[r * K_CODES + c];
        atomicAdd(avgAcc + c, a);
    }
}

// -------------------------------------------------------------------------
// Kernel 2: finalize scalars.
// -------------------------------------------------------------------------
__global__ void __launch_bounds__(1024)
vq_finalize(const float* __restrict__ avgAcc, const float* __restrict__ counts,
            const float* __restrict__ lossAcc, float* __restrict__ outScal,
            float invB, float invBD)
{
    __shared__ float s1[1024];
    __shared__ float s2[1024];
    const int t = threadIdx.x;
    const float avg = avgAcc[t] * invB;
    const float ha  = counts[t] * invB;
    s1[t] = -avg * __logf(avg + 1e-10f);
    s2[t] = -ha  * __logf(ha  + 1e-10f);
    __syncthreads();
    for (int off = 512; off > 0; off >>= 1) {
        if (t < off) { s1[t] += s1[t + off]; s2[t] += s2[t + off]; }
        __syncthreads();
    }
    if (t == 0) {
        const float usage  = s1[0];
        const float commit = lossAcc[0] * invBD;
        outScal[0] = commit;                       // commitment_loss
        outScal[1] = commit;                       // codebook_loss (== numerically)
        outScal[2] = -usage / logf(1024.0f);       // diversity_loss
        outScal[3] = usage;                        // usage_entropy
        outScal[4] = __expf(s2[0]);                // perplexity
    }
}

// -------------------------------------------------------------------------
extern "C" void kernel_launch(void* const* d_in, const int* in_sizes, int n_in,
                              void* d_out, int out_size, void* d_ws, size_t ws_size,
                              hipStream_t stream)
{
    (void)n_in; (void)out_size; (void)ws_size;
    const float* z     = (const float*)d_in[0];
    const float* embed = (const float*)d_in[1];
    const int B = in_sizes[0] / DIMS;              // 65536

    // workspace layout
    char* ws = (char*)d_ws;
    __bf16* ebf    = (__bf16*)ws;                      // 1024*256*2 = 524288
    float*  enorm  = (float*)(ws + 524288);            // 4096
    float*  avgAcc = (float*)(ws + 528384);            // 4096
    float*  counts = (float*)(ws + 532480);            // 4096
    float*  lossAc = (float*)(ws + 536576);            // 16

    // zero accumulators every call (harness does not re-poison/zero)
    hipMemsetAsync(ws + 528384, 0, 4096 + 4096 + 16, stream);

    float* out_zq    = (float*)d_out;
    float* out_codes = out_zq + (size_t)B * DIMS;
    float* out_scal  = out_codes + B;

    vq_prep<<<K_CODES / 8, 256, 0, stream>>>(embed, ebf, enorm);
    vq_main<<<B / ROWS_PB, 256, SMEM_BYTES, stream>>>(
        z, embed, ebf, enorm, avgAcc, counts, lossAc, out_zq, out_codes);
    vq_finalize<<<1, 1024, 0, stream>>>(avgAcc, counts, lossAc, out_scal,
                                        1.0f / (float)B,
                                        1.0f / ((float)B * (float)DIMS));
}
